// Network_SNN_22393959481550
// MI455X (gfx1250) — matrix operand order repaired and checked by reading, not compiled, not run on hardware
//
#include <hip/hip_runtime.h>

// ---------------------------------------------------------------------------
// SNN forward, MI455X (gfx1250, wave32, WMMA + TDM).
// One workgroup (8 wave32) per image; state LDS-resident for all 35 steps.
// Convs = implicit GEMM on v_wmma_f32_16x16x32_f16.
//  - w2/w3 pre-swizzled per-lane-contiguous A fragments in global scratch
//    (read-only, 77.8 KB -> WGP$-resident; A operand = 2x global_load_b128)
//  - spike maps channel-last: B operand = 2x ds_load_b128
//  - membranes channel-last: LIF = b128 load/store pairs
//  - branch-free clamped gathers (A zero-padding makes pad B values don't-care)
//  - LDS total 162,048 B <= 160 KiB -> 2 workgroups per WGP (4 waves/SIMD)
// ---------------------------------------------------------------------------

typedef __attribute__((ext_vector_type(16))) _Float16 v16h;
typedef __attribute__((ext_vector_type(8)))  _Float16 v8h;
typedef __attribute__((ext_vector_type(8)))  float    v8f;
typedef __attribute__((ext_vector_type(4)))  float    v4f;
typedef __attribute__((ext_vector_type(4)))  unsigned tdm_u4;
typedef __attribute__((ext_vector_type(4)))  int      tdm_i4;
typedef __attribute__((ext_vector_type(8)))  int      tdm_i8;

#define TIME_WINDOW 35
#define NTHREADS 256

// f32 LDS element counts (channel-last)
#define M1N (28*28*16)
#define P1N (14*14*16)
#define M2N (14*14*32)
#define P2N (7*7*32)
#define M3N (16*32)
#define P3N (4*32)

// f16 LDS element counts (multiples of 8 -> 16B-aligned bases)
#define SPKIN_N (32*32)
#define S1N   (28*28*16)
#define SP1N  (18*18*16)
#define S2N   (14*14*32)
#define SP2N  (11*11*32)
#define S3N   (16*32)
#define SP3N  (4*32)
#define W1FN  (1*512)
// global-scratch fragment counts
#define W2FN  (26*512)
#define W3FN  (50*512)

#define F32_CNT (M1N+P1N+M2N+P2N+M3N+M3N+P3N+16+16+784)
#define F16_CNT (SPKIN_N+S1N+SP1N+S2N+SP2N+S3N+SP3N+W1FN)
#define SMEM_BYTES (F32_CNT*4 + F16_CNT*2)        // 162,048 B -> 2 WG/WGP
#define XL_BYTE_OFF ((F32_CNT-784)*4)

// A 16x32 fragment: lane<16: M=lane, half i -> K=i; lane>=16: K=16+i
__device__ __forceinline__ int a_k_of(int lane, int i) {
  return i + ((lane & 16) ? 16 : 0);
}
// B 32x16 fragment: lane<16: N=lane, K=i; lane>=16: N=lane-16, K=16+i
// C/D 16x16 f32: element v -> M = v + (lane&16 ? 8 : 0), N = lane&15

__device__ __forceinline__ v8f wmma_f16(v16h a, v16h b, v8f c) {
  return __builtin_amdgcn_wmma_f32_16x16x32_f16(
      false, a, false, b, (short)0, c, false, false);
}

union V16 { v16h v; v8h q[2]; _Float16 h[16]; };
union V8  { v8f  v; float f[8]; };
union V8H { v8h  v; _Float16 h[8]; };

__device__ __forceinline__ float rand01(unsigned s) {
  s ^= s >> 17; s *= 0xed5ad4bbu;
  s ^= s >> 11; s *= 0xac4c1b51u;
  s ^= s >> 15; s *= 0x31848babu;
  s ^= s >> 14;
  return (float)(s >> 8) * (1.0f / 16777216.0f);
}

// 8-wide LIF against channel-last membrane (LDS) + spike (LDS) arrays.
__device__ __forceinline__ void lif8(float* __restrict__ memp,
                                     _Float16* __restrict__ spkp,
                                     const V8& c) {
  v4f mlo = *(v4f*)memp;
  v4f mhi = *(v4f*)(memp + 4);
  V8H s;
#pragma unroll
  for (int v = 0; v < 4; ++v) {
    float mm = mlo[v] + c.f[v];
    bool sp = (mm >= 1.0f);
    mlo[v] = sp ? 0.0f : mm;
    s.h[v] = (_Float16)(sp ? 1.0f : 0.0f);
  }
#pragma unroll
  for (int v = 0; v < 4; ++v) {
    float mm = mhi[v] + c.f[4 + v];
    bool sp = (mm >= 1.0f);
    mhi[v] = sp ? 0.0f : mm;
    s.h[4 + v] = (_Float16)(sp ? 1.0f : 0.0f);
  }
  *(v4f*)memp = mlo;
  *(v4f*)(memp + 4) = mhi;
  *(v8h*)spkp = s.v;
}

// 2x2 avg-pool over 8 contiguous channels + LIF; all vector LDS accesses.
__device__ __forceinline__ void pool_lif8(const _Float16* __restrict__ s00,
                                          int dx, int dyrow,
                                          float* __restrict__ memp,
                                          _Float16* __restrict__ spkp) {
  v8h a = *(const v8h*)s00;
  v8h b = *(const v8h*)(s00 + dx);
  v8h c = *(const v8h*)(s00 + dyrow);
  v8h d = *(const v8h*)(s00 + dyrow + dx);
  V8H su; su.v = a + b + c + d;            // v_pk_add_f16, exact (0..4)
  v4f mlo = *(v4f*)memp;
  v4f mhi = *(v4f*)(memp + 4);
  V8H out;
#pragma unroll
  for (int k = 0; k < 4; ++k) {
    float mm = mlo[k] + 0.25f * (float)su.h[k];
    bool sp = (mm >= 1.0f);
    mlo[k] = sp ? 0.0f : mm;
    out.h[k] = (_Float16)(sp ? 1.0f : 0.0f);
  }
#pragma unroll
  for (int k = 0; k < 4; ++k) {
    float mm = mhi[k] + 0.25f * (float)su.h[4 + k];
    bool sp = (mm >= 1.0f);
    mhi[k] = sp ? 0.0f : mm;
    out.h[4 + k] = (_Float16)(sp ? 1.0f : 0.0f);
  }
  *(v4f*)memp = mlo;
  *(v4f*)(memp + 4) = mhi;
  *(v8h*)spkp = out.v;
}

// ---- setup: pre-swizzle w2+w3 A-fragments into global scratch -------------
__global__ void wswizzle_kernel(const float* __restrict__ gw2,
                                const float* __restrict__ gw3,
                                _Float16* __restrict__ wg) {
  int i = blockIdx.x * blockDim.x + threadIdx.x;
  if (i < W2FN) {
    int f = i >> 9, rem = i & 511, L = rem >> 4, e = rem & 15;
    int mt = f / 13, kt = f % 13;
    int m = mt * 16 + (L & 15);
    int kg = kt * 32 + a_k_of(L, e);
    int tap = kg >> 4, ch = kg & 15;
    wg[i] = (_Float16)(tap < 25 ? gw2[(m * 16 + ch) * 25 + tap] : 0.0f);
  } else if (i < W2FN + W3FN) {
    int i2 = i - W2FN;
    int f = i2 >> 9, rem = i2 & 511, L = rem >> 4, e = rem & 15;
    int mt = f / 25, tap = f % 25;
    int m = mt * 16 + (L & 15);
    int ch = a_k_of(L, e);
    wg[i] = (_Float16)gw3[(m * 32 + ch) * 25 + tap];
  }
}

__global__ void __launch_bounds__(NTHREADS)
snn_lds_wmma_kernel(const float* __restrict__ gx,
                    const float* __restrict__ gw1,
                    const _Float16* __restrict__ wg,   // w2frag | w3frag
                    const float* __restrict__ gwfc,
                    float* __restrict__ gout) {
  extern __shared__ char smem_raw[];
  const int tid  = threadIdx.x;
  const int lane = tid & 31;
  const int wave = tid >> 5;
  const int img  = blockIdx.x;
  const int base8 = (lane & 16) ? 8 : 0;
  const _Float16* w2g = wg;
  const _Float16* w3g = wg + W2FN;

  // ---- carve LDS ----
  float* m1   = (float*)smem_raw;            // [784][16]
  float* p1cl = m1 + M1N;                    // [196][16]
  float* m2   = p1cl + P1N;                  // [196][32]
  float* p2cl = m2 + M2N;                    // [49][32]
  float* m3i  = p2cl + P2N;                  // [16][32]
  float* m3   = m3i + M3N;                   // [16][32]
  float* p3cl = m3 + M3N;                    // [4][32]
  float* mf   = p3cl + P3N;
  float* ssum = mf + 16;
  float* xl   = ssum + 16;                   // 784
  _Float16* spkin  = (_Float16*)(xl + 784);  // [32][32] padded
  _Float16* s1cl   = spkin + SPKIN_N;        // [784][16]
  _Float16* sp1cl  = s1cl + S1N;             // [18][18][16]
  _Float16* s2cl   = sp1cl + SP1N;           // [196][32]
  _Float16* sp2cl  = s2cl + S2N;             // [11][11][32]
  _Float16* s3cl   = sp2cl + SP2N;           // [16][32]
  _Float16* sp3cl  = s3cl + S3N;             // [4][32]
  _Float16* w1frag = sp3cl + SP3N;           // 512

  __builtin_prefetch(w2g, 0, 0);             // global_prefetch_b8
  __builtin_prefetch(w3g, 0, 0);

  // conv1 tap gather offsets: loop/time-invariant, branch-free (clamped;
  // A fragment is zero for K>=25 so clamped B values are don't-care).
  int boff[16];
#pragma unroll
  for (int i = 0; i < 16; ++i) {
    int kb = i + ((lane & 16) ? 16 : 0);
    int kbc = kb < 25 ? kb : 24;
    boff[i] = (kbc / 5) * 32 + (kbc % 5);
  }

  // ---- init: zero LDS (pads stay 0 forever) ----
  for (int i = tid; i < F32_CNT; i += NTHREADS) ((float*)smem_raw)[i] = 0.0f;
  for (int i = tid; i < F16_CNT; i += NTHREADS) spkin[i] = (_Float16)0.0f;
  __syncthreads();

  // ---- TDM: DMA the image (784 f32) HBM -> LDS xl (ISA 8.3/8.4 D#) ----
#if __has_builtin(__builtin_amdgcn_tensor_load_to_lds)
  if (wave == 0) {
    unsigned long long ga = (unsigned long long)(const void*)(gx + (size_t)img * 784);
    tdm_u4 g0;
    g0.x = 1u;                                          // count=1, user mode
    g0.y = (unsigned)XL_BYTE_OFF;                       // lds_addr
    g0.z = (unsigned)(ga & 0xffffffffu);                // global_addr lo
    g0.w = (unsigned)((ga >> 32) & 0x1ffffffu) | (2u << 30);  // hi | type=2
    tdm_i8 g1;
    g1[0] = (int)(2u << 16);     // data_size = 4B
    g1[1] = (int)(784u << 16);   // tensor_dim0 lo16
    g1[2] = (int)(1u << 16);     // tensor_dim1 = 1
    g1[3] = (int)(784u << 16);   // tile_dim0 = 784
    g1[4] = 1;                   // tile_dim1 = 1
    g1[5] = 784;                 // tensor_dim0_stride lo32
    g1[6] = (int)(784u << 16);   // tensor_dim1_stride lo16
    g1[7] = 0;
    tdm_i4 gz = {0, 0, 0, 0};
#if defined(__clang_major__) && (__clang_major__ >= 23)
    tdm_i8 gz8 = {0, 0, 0, 0, 0, 0, 0, 0};
    __builtin_amdgcn_tensor_load_to_lds(g0, g1, gz, gz, gz8, 0);
#else
    __builtin_amdgcn_tensor_load_to_lds(g0, g1, gz, gz, 0);
#endif
    __builtin_amdgcn_s_wait_tensorcnt((short)0);
  }
#else
  for (int i = tid; i < 784; i += NTHREADS) xl[i] = gx[img * 784 + i];
#endif

  // ---- w1 fragment pre-swizzled into LDS ----
  for (int i = tid; i < W1FN; i += NTHREADS) {
    int L = i >> 4, e = i & 15;
    int k = a_k_of(L, e);
    w1frag[i] = (_Float16)(k < 25 ? gw1[(L & 15) * 25 + k] : 0.0f);
  }
  __syncthreads();

  // ================= time loop =================
  for (int t = 0; t < TIME_WINDOW; ++t) {
    // -- stage 0: input spikes -> padded 32x32 map
    for (int p = tid; p < 784; p += NTHREADS) {
      unsigned seed = (unsigned)(img * 784 + p) * 2654435761u ^ ((unsigned)t * 0x9E3779B9u);
      float spk = (rand01(seed) * 5.0f <= xl[p]) ? 1.0f : 0.0f;
      spkin[(p / 28 + 2) * 32 + (p % 28 + 2)] = (_Float16)spk;
    }
    __syncthreads();

    // -- stage 1: conv1 WMMA. M=16ch, K=25(pad32), N=784 (49 tiles), rolled
    {
      V16 a1;
      const v8h* ap = (const v8h*)(w1frag + lane * 16);
      a1.q[0] = ap[0]; a1.q[1] = ap[1];
      for (int nt = wave; nt < 49; nt += 8) {
        const int pos = nt * 16 + (lane & 15);
        const int sbase = (pos / 28) * 32 + (pos % 28);
        V16 b; V8 c;
#pragma unroll
        for (int i = 0; i < 16; ++i)          // unconditional clamped gather
          b.h[i] = spkin[sbase + boff[i]];
#pragma unroll
        for (int i = 0; i < 8; ++i) c.f[i] = 0.0f;
        c.v = wmma_f16(a1.v, b.v, c.v);
        lif8(m1 + pos * 16 + base8, s1cl + pos * 16 + base8, c);
      }
    }
    __syncthreads();

    // -- stage 2: pool1 + LIF, vectorized over 8-channel groups (392 items)
    for (int i = tid; i < 392; i += NTHREADS) {
      int pos = i >> 1, g = (i & 1) * 8;
      int r = pos / 14, cc = pos % 14;
      pool_lif8(s1cl + ((2 * r) * 28 + 2 * cc) * 16 + g, 16, 28 * 16,
                p1cl + pos * 16 + g,
                sp1cl + ((r + 2) * 18 + (cc + 2)) * 16 + g);
    }
    __syncthreads();

    // -- stage 3: conv2 WMMA. M=32 (2 tiles), N=196 (13 tiles), K=400 (13x32)
    for (int ot = wave; ot < 26; ot += 8) {
      const int mt = ot / 13, nt = ot % 13;
      const int pos = nt * 16 + (lane & 15);
      const int posc = pos < 196 ? pos : 195;
      const int oy = posc / 14, ox = posc % 14;
      V8 c;
#pragma unroll
      for (int i = 0; i < 8; ++i) c.f[i] = 0.0f;
#pragma unroll
      for (int kt = 0; kt < 13; ++kt) {
        V16 a, b;
        const v8h* ap = (const v8h*)(w2g + (mt * 13 + kt) * 512 + lane * 16);
        a.q[0] = ap[0]; a.q[1] = ap[1];      // global_load_b128 x2 (WGP$ hit)
        int tap = 2 * kt + ((lane >> 4) & 1);
        int tapc = tap < 25 ? tap : 24;      // branch-free (A zero-padded)
        const v8h* bp = (const v8h*)(sp1cl + ((oy + tapc / 5) * 18 + ox + tapc % 5) * 16);
        b.q[0] = bp[0]; b.q[1] = bp[1];
        c.v = wmma_f16(a.v, b.v, c.v);
      }
      if (pos < 196) {
        int off = pos * 32 + mt * 16 + base8;
        lif8(m2 + off, s2cl + off, c);
      }
    }
    __syncthreads();

    // -- stage 4: pool2 + LIF vectorized (196 items); clear conv3 partials
    for (int i = tid; i < 196; i += NTHREADS) {
      int pos = i >> 2, g = (i & 3) * 8;
      int r = pos / 7, cc = pos % 7;
      pool_lif8(s2cl + ((2 * r) * 14 + 2 * cc) * 32 + g, 32, 14 * 32,
                p2cl + pos * 32 + g,
                sp2cl + ((r + 2) * 11 + (cc + 2)) * 32 + g);
    }
    for (int i = tid; i < M3N; i += NTHREADS) m3i[i] = 0.0f;
    __syncthreads();

    // -- stage 5: conv3 (stride2) WMMA; A from global scratch (WGP$-resident);
    //    taps split across wave-pairs, ds_add_f32 partial reduction.
    {
      const int mt = wave & 1;
      const int chunk = wave >> 1;
      const int tstart = chunk * 6 + (chunk > 0 ? 1 : 0);
      const int tcount = (chunk == 0) ? 7 : 6;
      const int n = lane & 15;
      const int oy = n / 4, ox = n % 4;
      V8 c;
#pragma unroll
      for (int i = 0; i < 8; ++i) c.f[i] = 0.0f;
      for (int j = 0; j < tcount; ++j) {
        const int tap = tstart + j;
        const int dy = tap / 5, dx = tap % 5;
        V16 a, b;
        const v8h* ap = (const v8h*)(w3g + (mt * 25 + tap) * 512 + lane * 16);
        a.q[0] = ap[0]; a.q[1] = ap[1];
        const v8h* bp = (const v8h*)(sp2cl + ((2 * oy + dy) * 11 + 2 * ox + dx) * 32
                                     + ((lane & 16) ? 16 : 0));
        b.q[0] = bp[0]; b.q[1] = bp[1];
        c.v = wmma_f16(a.v, b.v, c.v);
      }
#pragma unroll
      for (int v = 0; v < 8; ++v)
        atomicAdd(&m3i[n * 32 + mt * 16 + base8 + v], c.f[v]);   // ds_add_f32
    }
    __syncthreads();

    // -- stage 6a: conv3 LIF (channel-last [pos][32])
    for (int i = tid; i < M3N; i += NTHREADS) {
      float mm = m3[i] + m3i[i];
      bool sp = (mm >= 1.0f);
      m3[i] = sp ? 0.0f : mm;
      s3cl[i] = (_Float16)(sp ? 1.0f : 0.0f);
    }
    __syncthreads();
    // -- stage 6b: pool3 + LIF vectorized (16 items), channel-last sp3
    for (int i = tid; i < 16; i += NTHREADS) {
      int pos = i >> 2, g = (i & 3) * 8;
      int r = pos >> 1, cc = pos & 1;
      pool_lif8(s3cl + ((2 * r) * 4 + 2 * cc) * 32 + g, 32, 4 * 32,
                p3cl + pos * 32 + g,
                sp3cl + pos * 32 + g);
    }
    __syncthreads();
    // -- stage 6c: fc (10x128) + LIF + spike accumulation.
    //    reference flatten order: j = ch*4 + pos; sp3cl is [pos][ch].
    if (tid < 10) {
      float acc = 0.0f;
      for (int ch = 0; ch < 32; ++ch)
        for (int pos = 0; pos < 4; ++pos)
          acc += gwfc[tid * 128 + ch * 4 + pos] * (float)sp3cl[pos * 32 + ch];
      float mm = mf[tid] + acc;
      bool sp = (mm >= 1.0f);
      mf[tid] = sp ? 0.0f : mm;
      ssum[tid] += sp ? 1.0f : 0.0f;
    }
    __syncthreads();
  }

  if (tid < 10) gout[img * 10 + tid] = ssum[tid] * (1.0f / (float)TIME_WINDOW);
}

extern "C" void kernel_launch(void* const* d_in, const int* in_sizes, int n_in,
                              void* d_out, int out_size, void* d_ws, size_t ws_size,
                              hipStream_t stream) {
  (void)n_in; (void)out_size; (void)ws_size;
  const float* x   = (const float*)d_in[0];
  const float* w1  = (const float*)d_in[1];
  const float* w2  = (const float*)d_in[2];
  const float* w3  = (const float*)d_in[3];
  const float* wfc = (const float*)d_in[4];
  float* out = (float*)d_out;
  _Float16* wg = (_Float16*)d_ws;           // (26+50)*512 f16 = 77.8 KB
  const int B = in_sizes[0] / 784;          // 4096

  wswizzle_kernel<<<(W2FN + W3FN + NTHREADS - 1) / NTHREADS, NTHREADS, 0, stream>>>(w2, w3, wg);

  hipFuncSetAttribute(reinterpret_cast<const void*>(snn_lds_wmma_kernel),
                      hipFuncAttributeMaxDynamicSharedMemorySize, SMEM_BYTES);
  snn_lds_wmma_kernel<<<B, NTHREADS, SMEM_BYTES, stream>>>(x, w1, wg, wfc, out);
}
